// RecomNet_5901285064987
// MI455X (gfx1250) — matrix operand order repaired and checked by reading, not compiled, not run on hardware
//
#include <hip/hip_runtime.h>
#include <stdint.h>

// ---------------------------------------------------------------------------
// RecomNet on MI455X (gfx1250): bf16 split-precision WMMA GEMMs + L2-resident
// edge scatter/gather.  N=8192 nodes, E=524288 edges/type, H=200.
// ---------------------------------------------------------------------------

typedef __attribute__((ext_vector_type(16))) __bf16 v16bf;
typedef __attribute__((ext_vector_type(8)))  float  v8f;
typedef __attribute__((ext_vector_type(4)))  float  v4f;

#define NN   8192
#define HH   200
#define EE   524288
#define MEAN_RATING 3.5f
#define TPW  8        // n-tiles per wave (128 output cols per block)

// -------------------------------------------------------------------------
// Weight pre-swizzle: build hi/lo bf16 planes in the exact WMMA B-matrix
// register layout so the GEMM loads one contiguous 32B chunk per lane/tile.
//   B 32x16 bf16 layout: lane L -> col n = L%16, kbase = (L/16)*8,
//   element i (0..15) of the per-lane v16bf  ->  K = kbase + (i&7) + (i>>3)*16.
// Storage: [kblock][ntile][lane][i], columns = concat of up to 3 [K,Hm] mats.
// Tiles beyond the real column count (padding to a multiple of TPW) and
// K rows beyond Ktrue are zero-filled, so the GEMM needs no inner branches.
// -------------------------------------------------------------------------
__global__ void convert_w_swz(const float* __restrict__ w0,
                              const float* __restrict__ w1,
                              const float* __restrict__ w2,
                              int Hm, int nmat, int Ktrue,
                              int kblocks, int nTilesPad,
                              __bf16* __restrict__ outHi,
                              __bf16* __restrict__ outLo)
{
    size_t total = (size_t)kblocks * nTilesPad * 512;
    for (size_t idx = (size_t)blockIdx.x * blockDim.x + threadIdx.x;
         idx < total; idx += (size_t)gridDim.x * blockDim.x) {
        int    i    = (int)(idx & 15);
        int    lane = (int)((idx >> 4) & 31);
        size_t tile = idx >> 9;
        int    nt   = (int)(tile % (size_t)nTilesPad);
        int    kb   = (int)(tile / (size_t)nTilesPad);
        int    k    = kb * 32 + ((lane >> 4) & 1) * 8 + (i & 7) + (i >> 3) * 16;
        int    col  = nt * 16 + (lane & 15);
        float  v    = 0.0f;
        int    msel = col / Hm;
        int    c    = col - msel * Hm;
        if (k < Ktrue && msel < nmat) {
            const float* w = (msel == 0) ? w0 : ((msel == 1) ? w1 : w2);
            v = w[(size_t)k * Hm + c];
        }
        __bf16 h = (__bf16)v;
        outHi[idx] = h;
        outLo[idx] = (__bf16)(v - (float)h);
    }
}

// -------------------------------------------------------------------------
// GEMM  Y[M,ncols] = A[M,Ktrue](fp32) * B  with B pre-split hi/lo bf16.
// Split-precision: A = Ahi + Alo (bf16 each);
//   acc += Ahi*Bhi + Ahi*Blo + Alo*Bhi   (3 WMMAs, fp32 accumulate)
// Block = 256 threads = 8 waves; wave w owns rows [blk.x*128 + 16w, +16) and
// TPW=8 n-tiles (128 cols) from blk.y.  K and tiles are zero-padded, so the
// inner loop is branch-free: 4x b128 A loads + 32x b128 B loads + 24 WMMAs.
// -------------------------------------------------------------------------
__global__ __launch_bounds__(256)
void gemm_bf16x3(const float* __restrict__ A, int lda, int Ktrue, int kblocks,
                 const __bf16* __restrict__ Bhi, const __bf16* __restrict__ Blo,
                 int nTilesPad, float* __restrict__ Y, int ldy, int ncols)
{
    const int lane   = threadIdx.x & 31;
    const int wave   = threadIdx.x >> 5;
    const int mbase  = blockIdx.x * 128 + wave * 16;
    const int ntile0 = blockIdx.y * TPW;

    // A-matrix 16x32 bf16 layout: lane L -> row M = L%16, kofs = (L/16)*8,
    // v16bf element e -> K = kofs + e (e<8)  |  16 + kofs + (e-8) (e>=8).
    const int m    = mbase + (lane & 15);
    const int kofs = ((lane >> 4) & 1) * 8;
    const float* arow = A + (size_t)m * lda;

    v8f acc[TPW];
    {
        v8f zero = {};
#pragma unroll
        for (int t = 0; t < TPW; ++t) acc[t] = zero;
    }

    for (int kb = 0; kb < kblocks; ++kb) {
        const int k0 = kb * 32 + kofs;       // first 8-float run
        const int k1 = k0 + 16;              // second 8-float run
        float a[16];
        if (k1 + 8 <= Ktrue) {               // fast path: 4 x b128
            v4f p0 = *(const v4f*)(arow + k0);
            v4f p1 = *(const v4f*)(arow + k0 + 4);
            v4f p2 = *(const v4f*)(arow + k1);
            v4f p3 = *(const v4f*)(arow + k1 + 4);
#pragma unroll
            for (int e = 0; e < 4; ++e) {
                a[e]      = p0[e]; a[4 + e]  = p1[e];
                a[8 + e]  = p2[e]; a[12 + e] = p3[e];
            }
        } else {                             // K tail: bounds-checked scalars
#pragma unroll
            for (int e = 0; e < 8; ++e) {
                a[e]     = (k0 + e < Ktrue) ? arow[k0 + e] : 0.0f;
                a[8 + e] = (k1 + e < Ktrue) ? arow[k1 + e] : 0.0f;
            }
        }
        // A-stream prefetch: a 128B line covers 1 K-block; pull 4 ahead.
        if (kb + 4 < kblocks) {
            __builtin_prefetch(arow + k0 + 128, 0, 0);   // global_prefetch_b8
            __builtin_prefetch(arow + k1 + 128, 0, 0);
        }

        v16bf ahi, alo;
#pragma unroll
        for (int e = 0; e < 16; ++e) {
            __bf16 h = (__bf16)a[e];
            ahi[e] = h;
            alo[e] = (__bf16)(a[e] - (float)h);
        }

        const __bf16* bh_p = Bhi + (((size_t)kb * nTilesPad + ntile0) * 32 + lane) * 16;
        const __bf16* bl_p = Blo + (((size_t)kb * nTilesPad + ntile0) * 32 + lane) * 16;
#pragma unroll
        for (int t = 0; t < TPW; ++t) {
            v16bf bh = *(const v16bf*)(bh_p + (size_t)t * 512);
            v16bf bl = *(const v16bf*)(bl_p + (size_t)t * 512);
            acc[t] = __builtin_amdgcn_wmma_f32_16x16x32_bf16(false, ahi, false, bh,
                                                             (short)0, acc[t], false, false);
            acc[t] = __builtin_amdgcn_wmma_f32_16x16x32_bf16(false, ahi, false, bl,
                                                             (short)0, acc[t], false, false);
            acc[t] = __builtin_amdgcn_wmma_f32_16x16x32_bf16(false, alo, false, bh,
                                                             (short)0, acc[t], false, false);
        }
    }

    // C/D 16x16 f32 layout: lane L -> col n = L%16; vgpr j -> row M = j + (L/16)*8
    const int rofs = ((lane >> 4) & 1) * 8;
#pragma unroll
    for (int t = 0; t < TPW; ++t) {
        const int col = (ntile0 + t) * 16 + (lane & 15);
        if (col < ncols) {
#pragma unroll
            for (int j = 0; j < 8; ++j) {
                Y[(size_t)(mbase + rofs + j) * ldy + col] = acc[t][j];
            }
        }
    }
}

// -------------------------------------------------------------------------
// Epilogues
// -------------------------------------------------------------------------
__global__ void init_bias(const float* __restrict__ Y, int ldy,
                          const float* __restrict__ b,
                          float* __restrict__ Hacc, int Hm, size_t total)
{
    size_t idx = (size_t)blockIdx.x * blockDim.x + threadIdx.x;
    if (idx >= total) return;
    size_t n = idx / Hm;
    int    h = (int)(idx - n * Hm);
    Hacc[idx] = Y[n * ldy + h] + b[h];
}

__global__ void scatter_add(const float* __restrict__ Y, int ldy, int colofs,
                            const int* __restrict__ esrc,
                            const int* __restrict__ edst,
                            float* __restrict__ Hacc, int Hm)
{
    const int e = blockIdx.x;
    const int h = threadIdx.x;
    if (h >= Hm) return;
    const int s = esrc[e];
    const int d = edst[e];
    atomicAdd(&Hacc[(size_t)d * Hm + h], Y[(size_t)s * ldy + colofs + h]);
}

__global__ void relu_inplace(float* __restrict__ X, size_t total)
{
    size_t idx = (size_t)blockIdx.x * blockDim.x + threadIdx.x;
    if (idx < total) X[idx] = fmaxf(X[idx], 0.0f);
}

// pred[e] = mask[e] ? dot(G[src[e]], H2[dst[e]]) + 3.5 : 0
__global__ void decode_edges(const float* __restrict__ G,
                             const float* __restrict__ H2,
                             const int* __restrict__ erat,
                             const uint8_t* __restrict__ mask,
                             float* __restrict__ out, int E, int Hm)
{
    int e = blockIdx.x * blockDim.x + threadIdx.x;
    if (e >= E) return;
    if (!mask[e]) { out[e] = 0.0f; return; }
    const int s = erat[e];
    const int d = erat[E + e];
    const float* g = G  + (size_t)s * Hm;
    const float* h = H2 + (size_t)d * Hm;
    float acc = 0.0f;
    for (int i = 0; i < Hm; i += 4) {
        v4f gv = *(const v4f*)(g + i);
        v4f hv = *(const v4f*)(h + i);
        acc += gv[0] * hv[0] + gv[1] * hv[1] + gv[2] * hv[2] + gv[3] * hv[3];
    }
    out[e] = acc + MEAN_RATING;
}

// -------------------------------------------------------------------------
static inline size_t alignup(size_t x) { return (x + 255) & ~(size_t)255; }

extern "C" void kernel_launch(void* const* d_in, const int* in_sizes, int n_in,
                              void* d_out, int out_size, void* d_ws, size_t ws_size,
                              hipStream_t stream)
{
    (void)in_sizes; (void)n_in; (void)out_size; (void)ws_size;

    const float*   x       = (const float*)  d_in[0];
    const int*     e_sim   = (const int*)    d_in[1];   // [2,E]
    const int*     e_rat   = (const int*)    d_in[2];   // [2,E]
    const uint8_t* mask    = (const uint8_t*)d_in[3];   // [E] bool
    const float*   w1_self = (const float*)  d_in[4];
    const float*   w1_sim  = (const float*)  d_in[5];
    const float*   w1_rat  = (const float*)  d_in[6];
    const float*   b1      = (const float*)  d_in[7];
    const float*   w2_self = (const float*)  d_in[8];
    const float*   w2_sim  = (const float*)  d_in[9];
    const float*   w2_rat  = (const float*)  d_in[10];
    const float*   b2      = (const float*)  d_in[11];
    const float*   Q       = (const float*)  d_in[12];
    float*         out     = (float*)        d_out;

    // ---- sizes ----
    const int KB1  = NN / 32;                          // 256 K-blocks, conv1
    const int KB2  = (HH + 31) / 32;                   // 7 K-blocks, K=200
    const int NT3  = ((3 * HH + 15) / 16 + TPW - 1) / TPW * TPW;  // 38 -> 40
    const int NTQ  = ((HH + 15) / 16 + TPW - 1) / TPW * TPW;      // 13 -> 16

    // ---- workspace carve-up ----
    char*  p   = (char*)d_ws;
    size_t ofs = 0;
    auto carve = [&](size_t bytes) { void* r = p + ofs; ofs = alignup(ofs + bytes); return r; };

    __bf16* W1hi = (__bf16*)carve((size_t)KB1 * NT3 * 512 * 2);
    __bf16* W1lo = (__bf16*)carve((size_t)KB1 * NT3 * 512 * 2);
    float*  Y1   = (float*) carve((size_t)NN * 600 * 4);
    float*  H1   = (float*) carve((size_t)NN * HH * 4);
    __bf16* W2hi = (__bf16*)carve((size_t)KB2 * NT3 * 512 * 2);
    __bf16* W2lo = (__bf16*)carve((size_t)KB2 * NT3 * 512 * 2);
    float*  Y2   = (float*) carve((size_t)NN * 600 * 4);
    float*  H2   = (float*) carve((size_t)NN * HH * 4);
    __bf16* Qhi  = (__bf16*)carve((size_t)KB2 * NTQ * 512 * 2);
    __bf16* Qlo  = (__bf16*)carve((size_t)KB2 * NTQ * 512 * 2);
    float*  G    = (float*) carve((size_t)NN * HH * 4);

    const dim3 blk256(256);
    const size_t nodeTot = (size_t)NN * HH;
    const int nodeBlks = (int)((nodeTot + 255) / 256);

    // ===== conv1 =====
    convert_w_swz<<<dim3(2048), blk256, 0, stream>>>(
        w1_self, w1_sim, w1_rat, HH, 3, NN, KB1, NT3, W1hi, W1lo);
    gemm_bf16x3<<<dim3(NN / 128, NT3 / TPW), blk256, 0, stream>>>(
        x, NN, NN, KB1, W1hi, W1lo, NT3, Y1, 600, 600);
    init_bias<<<dim3(nodeBlks), blk256, 0, stream>>>(Y1, 600, b1, H1, HH, nodeTot);
    scatter_add<<<dim3(EE), blk256, 0, stream>>>(Y1, 600, HH,     e_sim, e_sim + EE, H1, HH);
    scatter_add<<<dim3(EE), blk256, 0, stream>>>(Y1, 600, 2 * HH, e_rat, e_rat + EE, H1, HH);
    relu_inplace<<<dim3(nodeBlks), blk256, 0, stream>>>(H1, nodeTot);

    // ===== conv2 =====
    convert_w_swz<<<dim3(256), blk256, 0, stream>>>(
        w2_self, w2_sim, w2_rat, HH, 3, HH, KB2, NT3, W2hi, W2lo);
    gemm_bf16x3<<<dim3(NN / 128, NT3 / TPW), blk256, 0, stream>>>(
        H1, HH, HH, KB2, W2hi, W2lo, NT3, Y2, 600, 600);
    init_bias<<<dim3(nodeBlks), blk256, 0, stream>>>(Y2, 600, b2, H2, HH, nodeTot);
    scatter_add<<<dim3(EE), blk256, 0, stream>>>(Y2, 600, HH,     e_sim, e_sim + EE, H2, HH);
    scatter_add<<<dim3(EE), blk256, 0, stream>>>(Y2, 600, 2 * HH, e_rat, e_rat + EE, H2, HH);

    // ===== decoder: G = H2 @ Q, then per-edge bilinear dot =====
    convert_w_swz<<<dim3(128), blk256, 0, stream>>>(
        Q, Q, Q, HH, 1, HH, KB2, NTQ, Qhi, Qlo);
    gemm_bf16x3<<<dim3(NN / 128, NTQ / TPW), blk256, 0, stream>>>(
        H2, HH, HH, KB2, Qhi, Qlo, NTQ, G, HH, HH);
    decode_edges<<<dim3(EE / 256), blk256, 0, stream>>>(G, H2, e_rat, mask, out, EE, HH);
}
